// SuperGlue_29059748724908
// MI455X (gfx1250) — matrix operand order repaired and checked by reading, not compiled, not run on hardware
//
#include <hip/hip_runtime.h>
#include <hip/hip_bf16.h>

// ---------------- constants ----------------
#define Bc   4
#define Nc   512
#define Dc   256
#define Hc   4
#define DHc  64
#define Lc   18
#define NP1  513
#define LOG1024f 6.93147180559945f   // log(M+N) = log(1024)
#define LN2f     0.69314718055995f

typedef _Float16 half16 __attribute__((ext_vector_type(16)));
typedef float    float8 __attribute__((ext_vector_type(8)));

union H16 { half16 v; _Float16 e[16]; };
union F8  { float8 v; float     e[8]; };

__device__ __forceinline__ void cvt4(const float4 v, _Float16* e) {
  e[0] = (_Float16)v.x; e[1] = (_Float16)v.y;
  e[2] = (_Float16)v.z; e[3] = (_Float16)v.w;
}

// A fragment (16-bit A 16x32, ISA 7.12.2): lane (hs,r) holds row m=r,
// K = k0+hs*8+0..7 and k0+16+hs*8+0..7  -> two contiguous 8-float spans.
__device__ __forceinline__ void load_A(const float* __restrict__ row, int k0, int hs, H16& f) {
  const float* p = row + k0 + hs * 8;
  cvt4(*(const float4*)(p),      f.e + 0);
  cvt4(*(const float4*)(p + 4),  f.e + 4);
  cvt4(*(const float4*)(p + 16), f.e + 8);
  cvt4(*(const float4*)(p + 20), f.e + 12);
}

// B fragment (16-bit B 32x16): lane (hs,r) holds col r,
// K = k0+hs*8+0..7 and k0+16+hs*8+0..7  -> two contiguous 8-float spans.
__device__ __forceinline__ void load_B(const float* __restrict__ row, int k0, int hs, H16& f) {
  const float* p = row + k0 + hs * 8;
  cvt4(*(const float4*)(p),      f.e + 0);
  cvt4(*(const float4*)(p + 4),  f.e + 4);
  cvt4(*(const float4*)(p + 16), f.e + 8);
  cvt4(*(const float4*)(p + 20), f.e + 12);
}

// =====================================================================
// Generic conv1 GEMM on transposed activations.
//   acc[m, n] = sum_k W[m,k] * Xt[n,k]      (Xt row stride = K)
// Store modes:
//   0: Yt[n, m] = acc + bias                (yld = out row stride)
//   1: Yt[n, m] = ReLU(BN(acc + bias))
//   2: Yt[n, m] = acc + bias + Rt[n, m]
//   3: q/k head-split store: Q[(bz*H + c&3)][n][c>>2] (c = channel)
//   4: (c, n) store: Ycn[m, n] = acc + bias (for V)
// One wave: 16(M) x 64(N) strip; A loaded once per K-step, 4 WMMAs.
// =====================================================================
__global__ void gemm_T(const float* __restrict__ W, const float* __restrict__ X,
                       const float* __restrict__ bias,
                       const float* __restrict__ g,  const float* __restrict__ be,
                       const float* __restrict__ mu, const float* __restrict__ var,
                       const float* __restrict__ res,
                       float* __restrict__ Y,
                       int M, int K, int yld, long xbs, long ybs, long rbs, int mode)
{
  const int lane = threadIdx.x & 31;
  const int hs   = lane >> 4;
  const int r    = lane & 15;
  const int n0   = blockIdx.x * 64;
  const int m0   = blockIdx.y * 16;
  const int bz   = blockIdx.z;
  const float* Xb = X + (size_t)bz * xbs;
  const float* wrow = W + (size_t)(m0 + r) * K;
  const float* xrow0 = Xb + (size_t)(n0 + r) * K;

  F8 acc[4];
#pragma unroll
  for (int j = 0; j < 4; ++j)
#pragma unroll
    for (int t = 0; t < 8; ++t) acc[j].e[t] = 0.f;

  for (int k0 = 0; k0 < K; k0 += 32) {
    if (k0 + 64 < K) {
      __builtin_prefetch(wrow + k0 + 64, 0, 1);
      __builtin_prefetch(xrow0 + k0 + 64, 0, 1);
    }
    H16 a;
    load_A(wrow, k0, hs, a);
#pragma unroll
    for (int j = 0; j < 4; ++j) {
      H16 b;
      load_B(Xb + (size_t)(n0 + 16 * j + r) * K, k0, hs, b);
      acc[j].v = __builtin_amdgcn_wmma_f32_16x16x32_f16(false, a.v, false, b.v,
                                                        (short)0, acc[j].v, false, false);
    }
  }

  // per-row (channel) epilogue params
  float bsv[8], sclv[8], subv[8], addv[8];
#pragma unroll
  for (int t = 0; t < 8; ++t) {
    const int m = m0 + t + hs * 8;
    bsv[t] = bias[m];
    if (mode == 1) {
      sclv[t] = g[m] * rsqrtf(var[m] + 1e-5f);
      subv[t] = mu[m];
      addv[t] = be[m];
    }
  }

  if (mode == 3) {                       // q/k head-split: [bz*H + h][n][dh]
#pragma unroll
    for (int j = 0; j < 4; ++j) {
      const int col = n0 + 16 * j + r;
#pragma unroll
      for (int t = 0; t < 8; ++t) {
        const int c = m0 + t + hs * 8;
        Y[((size_t)bz * Hc + (c & 3)) * ((size_t)Nc * DHc) +
          (size_t)col * DHc + (c >> 2)] = acc[j].e[t] + bsv[t];
      }
    }
  } else if (mode == 4) {                // (c, n) layout for V
    float* Yb = Y + (size_t)bz * ybs;
#pragma unroll
    for (int j = 0; j < 4; ++j) {
      const int col = n0 + 16 * j + r;
#pragma unroll
      for (int t = 0; t < 8; ++t) {
        const int m = m0 + t + hs * 8;
        Yb[(size_t)m * Nc + col] = acc[j].e[t] + bsv[t];
      }
    }
  } else {                               // transposed (n, m) store, contiguous per lane
    float* Yb = Y + (size_t)bz * ybs;
#pragma unroll
    for (int j = 0; j < 4; ++j) {
      const int col = n0 + 16 * j + r;
      float* yr = Yb + (size_t)col * yld + m0 + hs * 8;
#pragma unroll
      for (int t = 0; t < 8; ++t) {
        float val = acc[j].e[t] + bsv[t];
        if (mode == 1) {
          val = (val - subv[t]) * sclv[t] + addv[t];
          val = fmaxf(val, 0.f);
        } else if (mode == 2) {
          val += res[(size_t)bz * rbs + (size_t)col * yld + m0 + hs * 8 + t];
        }
        yr[t] = val;
      }
    }
  }
}

// =====================================================================
// Attention scores: S[z,n,m] = (1/8) * sum_dh Q[z,n,dh] * K[z,m,dh]
// Q,K in head-split [z][n][64] layout -> both fragments fully contiguous.
// =====================================================================
__global__ void attn_scores_wmma(const float* __restrict__ Q, const float* __restrict__ Kd,
                                 float* __restrict__ S)
{
  const int lane = threadIdx.x & 31, hs = lane >> 4, r = lane & 15;
  const int m0 = blockIdx.x * 64;          // key index
  const int n0 = blockIdx.y * 16;          // query index
  const int z = blockIdx.z;
  const float* Qh = Q  + (size_t)z * Nc * DHc;
  const float* Kh = Kd + (size_t)z * Nc * DHc;

  F8 acc[4];
#pragma unroll
  for (int j = 0; j < 4; ++j)
#pragma unroll
    for (int t = 0; t < 8; ++t) acc[j].e[t] = 0.f;

  for (int k0 = 0; k0 < DHc; k0 += 32) {
    H16 a;
    load_A(Qh + (size_t)(n0 + r) * DHc, k0, hs, a);
#pragma unroll
    for (int j = 0; j < 4; ++j) {
      H16 b;
      load_B(Kh + (size_t)(m0 + 16 * j + r) * DHc, k0, hs, b);
      acc[j].v = __builtin_amdgcn_wmma_f32_16x16x32_f16(false, a.v, false, b.v,
                                                        (short)0, acc[j].v, false, false);
    }
  }

  float* Sz = S + (size_t)z * Nc * Nc;
#pragma unroll
  for (int t = 0; t < 8; ++t) {
    const int n = n0 + t + hs * 8;
#pragma unroll
    for (int j = 0; j < 4; ++j)
      Sz[(size_t)n * Nc + m0 + 16 * j + r] = acc[j].e[t] * 0.125f;   // / sqrt(64)
  }
}

// Softmax over last axis: one block (128 thr) per row of length 512.
__global__ void softmax_rows(float* __restrict__ S)
{
  __shared__ float red[128];
  float* p = S + (size_t)blockIdx.x * Nc;
  const int t = threadIdx.x;
  float x[4];
  float mx = -3.402823e38f;
#pragma unroll
  for (int i = 0; i < 4; ++i) { x[i] = p[t + i * 128]; mx = fmaxf(mx, x[i]); }
  red[t] = mx; __syncthreads();
  for (int o = 64; o > 0; o >>= 1) { if (t < o) red[t] = fmaxf(red[t], red[t + o]); __syncthreads(); }
  mx = red[0]; __syncthreads();
  float s = 0.f;
#pragma unroll
  for (int i = 0; i < 4; ++i) { x[i] = __expf(x[i] - mx); s += x[i]; }
  red[t] = s; __syncthreads();
  for (int o = 64; o > 0; o >>= 1) { if (t < o) red[t] += red[t + o]; __syncthreads(); }
  const float inv = 1.f / red[0];
#pragma unroll
  for (int i = 0; i < 4; ++i) p[t + i * 128] = x[i] * inv;
}

// =====================================================================
// Attention output, directly in transposed layout:
//   Ot[b][n][dc*4+h] = sum_m P[z,n,m] * V[b][dc*4+h][m]
// A = P rows (m-contiguous), B = V rows (m-contiguous): all b128 loads.
// One wave: 16 n-rows x all 64 dh channels (4 WMMAs, A reused).
// =====================================================================
__global__ void attn_out_wmma(const float* __restrict__ P, const float* __restrict__ V,
                              float* __restrict__ O)
{
  const int lane = threadIdx.x & 31, hs = lane >> 4, r = lane & 15;
  const int n0 = blockIdx.x * 16;
  const int z = blockIdx.z, b = z >> 2, h = z & 3;
  const float* Pz = P + (size_t)z * Nc * Nc;
  const float* Vb = V + (size_t)b * Dc * Nc;

  F8 acc[4];
#pragma unroll
  for (int j = 0; j < 4; ++j)
#pragma unroll
    for (int t = 0; t < 8; ++t) acc[j].e[t] = 0.f;

  const float* prow0 = Pz + (size_t)(n0 + r) * Nc;
  for (int k0 = 0; k0 < Nc; k0 += 32) {
    if (k0 + 64 < Nc) __builtin_prefetch(prow0 + k0 + 64, 0, 1);
    H16 a;
    load_A(prow0, k0, hs, a);                       // A(row n, k=m)
#pragma unroll
    for (int j = 0; j < 4; ++j) {
      H16 bb;                                       // B(k=m, col=dc)
      load_B(Vb + (size_t)((16 * j + r) * 4 + h) * Nc, k0, hs, bb);
      acc[j].v = __builtin_amdgcn_wmma_f32_16x16x32_f16(false, a.v, false, bb.v,
                                                        (short)0, acc[j].v, false, false);
    }
  }

  float* Ob = O + (size_t)b * Nc * Dc;
#pragma unroll
  for (int j = 0; j < 4; ++j) {
    const int c = (16 * j + r) * 4 + h;
#pragma unroll
    for (int t = 0; t < 8; ++t) {
      const int n = n0 + t + hs * 8;
      Ob[(size_t)n * Dc + c] = acc[j].e[t];
    }
  }
}

// Final match scores: C[b,n,m] = (1/16) * sum_d p0t[n,d] p1t[m,d]; ldc = 513.
__global__ void final_scores_wmma(const float* __restrict__ P0, const float* __restrict__ P1,
                                  float* __restrict__ C)
{
  const int lane = threadIdx.x & 31, hs = lane >> 4, r = lane & 15;
  const int m0 = blockIdx.x * 64;
  const int n0 = blockIdx.y * 16;
  const int b  = blockIdx.z;
  const float* At = P0 + (size_t)b * Nc * Dc;
  const float* Bt = P1 + (size_t)b * Nc * Dc;

  F8 acc[4];
#pragma unroll
  for (int j = 0; j < 4; ++j)
#pragma unroll
    for (int t = 0; t < 8; ++t) acc[j].e[t] = 0.f;

  for (int k0 = 0; k0 < Dc; k0 += 32) {
    H16 a;
    load_A(At + (size_t)(n0 + r) * Dc, k0, hs, a);
#pragma unroll
    for (int j = 0; j < 4; ++j) {
      H16 bb;
      load_B(Bt + (size_t)(m0 + 16 * j + r) * Dc, k0, hs, bb);
      acc[j].v = __builtin_amdgcn_wmma_f32_16x16x32_f16(false, a.v, false, bb.v,
                                                        (short)0, acc[j].v, false, false);
    }
  }

  float* Cb = C + (size_t)b * NP1 * NP1;
#pragma unroll
  for (int t = 0; t < 8; ++t) {
    const int n = n0 + t + hs * 8;
#pragma unroll
    for (int j = 0; j < 4; ++j)
      Cb[(size_t)n * NP1 + m0 + 16 * j + r] = acc[j].e[t] * 0.0625f;   // / sqrt(256)
  }
}

// -------------------- keypoint encoder (MLP 3->32->64->128->256->256) -----
__device__ __forceinline__ void enc_layer(const float* W, const float* bias,
                                          const float* g, const float* be,
                                          const float* mu, const float* var,
                                          const float* in, float* out,
                                          int Cin, int Cout, bool act)
{
  const int t = threadIdx.x;
  if (t < Cout) {
    float s = bias[t];
    const float* wr = W + (size_t)t * Cin;
    for (int i = 0; i < Cin; ++i) s += wr[i] * in[i];
    if (act) {
      s = (s - mu[t]) * (g[t] * rsqrtf(var[t] + 1e-5f)) + be[t];
      s = fmaxf(s, 0.f);
    }
    out[t] = s;
  }
  __syncthreads();
}

// writes transposed state:  xt[(b*Nc+n)*Dc + c] = desc[b][c][n] + enc[c]
__global__ void encoder_kernel(const float* __restrict__ kp, const float* __restrict__ sc,
                               const float* __restrict__ desc, float* __restrict__ out,
    const float* W0,const float* c0,const float* g0,const float* e0,const float* u0,const float* q0,
    const float* W1,const float* c1,const float* g1,const float* e1,const float* u1,const float* q1,
    const float* W2,const float* c2,const float* g2,const float* e2,const float* u2,const float* q2,
    const float* W3,const float* c3,const float* g3,const float* e3,const float* u3,const float* q3,
    const float* W4,const float* c4)
{
  __shared__ float bufA[256], bufB[256];
  const int p = blockIdx.x, b = p / Nc, n = p % Nc;
  if (threadIdx.x < 3) {
    bufA[threadIdx.x] = (threadIdx.x == 0) ? kp[(size_t)(b * Nc + n) * 2 + 0]
                      : (threadIdx.x == 1) ? kp[(size_t)(b * Nc + n) * 2 + 1]
                                           : sc[b * Nc + n];
  }
  __syncthreads();
  enc_layer(W0, c0, g0, e0, u0, q0, bufA, bufB,   3,  32, true);
  enc_layer(W1, c1, g1, e1, u1, q1, bufB, bufA,  32,  64, true);
  enc_layer(W2, c2, g2, e2, u2, q2, bufA, bufB,  64, 128, true);
  enc_layer(W3, c3, g3, e3, u3, q3, bufB, bufA, 128, 256, true);
  enc_layer(W4, c4, nullptr, nullptr, nullptr, nullptr, bufA, bufB, 256, 256, false);
  const int c = threadIdx.x;
  out[((size_t)b * Nc + n) * Dc + c] =
      desc[(size_t)b * Dc * Nc + (size_t)c * Nc + n] + bufB[c];
}

// copy xt (n,256) into first 256 columns of Cat_t (n,512)
__global__ void copy_to_cat(const float* __restrict__ X, float* __restrict__ Cat)
{
  const size_t i = (size_t)blockIdx.x * blockDim.x + threadIdx.x;
  if (i >= (size_t)Bc * Nc * Dc) return;
  const size_t row = i / Dc, c = i % Dc;
  Cat[row * (size_t)(2 * Dc) + c] = X[i];
}

// fill dustbin row/col with bin_score
__global__ void bins_kernel(float* __restrict__ C, const float* __restrict__ alpha)
{
  const int t = blockIdx.x * blockDim.x + threadIdx.x;
  if (t >= Bc * NP1) return;
  const float a = alpha[0];
  const int b = t / NP1, i = t % NP1;
  float* Cb = C + (size_t)b * NP1 * NP1;
  Cb[(size_t)512 * NP1 + i] = a;
  Cb[(size_t)i * NP1 + 512] = a;
}

// =====================================================================
// Fused Sinkhorn: 100 iterations, one block (16 waves) per batch.
// u, v live in LDS; online log-sum-exp + wave32 shuffle reduction.
// =====================================================================
__global__ void sinkhorn_kernel(const float* __restrict__ C,
                                float* __restrict__ uo, float* __restrict__ vo)
{
  __shared__ float su[NP1], sv[NP1];
  const int b = blockIdx.x;
  const float* Cb = C + (size_t)b * NP1 * NP1;
  const int tid = threadIdx.x, wave = tid >> 5, lane = tid & 31;
  const int NW = blockDim.x >> 5;
  for (int i = tid; i < NP1; i += blockDim.x) { su[i] = 0.f; sv[i] = 0.f; }
  __syncthreads();

  for (int it = 0; it < 100; ++it) {
    // u = log_mu - LSE_j(C[i,j] + v[j])
    for (int i = wave; i < NP1; i += NW) {
      const float* row = Cb + (size_t)i * NP1;
      float m = -3.402823e38f, s = 0.f;
      for (int j = lane; j < NP1; j += 32) {
        const float x = row[j] + sv[j];
        if (x <= m) s += __expf(x - m);
        else { s = s * __expf(m - x) + 1.f; m = x; }
      }
      for (int o = 16; o > 0; o >>= 1) {
        const float m2 = __shfl_xor(m, o, 32);
        const float s2 = __shfl_xor(s, o, 32);
        const float M = fmaxf(m, m2);
        s = s * __expf(m - M) + s2 * __expf(m2 - M);
        m = M;
      }
      if (lane == 0) {
        const float lmu = (i < 512) ? -LOG1024f : -LN2f;
        su[i] = lmu - (m + __logf(s));
      }
    }
    __syncthreads();
    // v = log_nu - LSE_i(C[i,j] + u[i])
    for (int j = wave; j < NP1; j += NW) {
      float m = -3.402823e38f, s = 0.f;
      for (int i = lane; i < NP1; i += 32) {
        const float x = Cb[(size_t)i * NP1 + j] + su[i];
        if (x <= m) s += __expf(x - m);
        else { s = s * __expf(m - x) + 1.f; m = x; }
      }
      for (int o = 16; o > 0; o >>= 1) {
        const float m2 = __shfl_xor(m, o, 32);
        const float s2 = __shfl_xor(s, o, 32);
        const float M = fmaxf(m, m2);
        s = s * __expf(m - M) + s2 * __expf(m2 - M);
        m = M;
      }
      if (lane == 0) {
        const float lnu = (j < 512) ? -LOG1024f : -LN2f;
        sv[j] = lnu - (m + __logf(s));
      }
    }
    __syncthreads();
  }
  for (int i = tid; i < NP1; i += blockDim.x) {
    uo[b * NP1 + i] = su[i];
    vo[b * NP1 + i] = sv[i];
  }
}

// row argmax over m of Z[b,n,m] (Z = C + u + v + log(1024)); keep value too.
__global__ void rowmax_kernel(const float* __restrict__ C, const float* __restrict__ u,
                              const float* __restrict__ v, int* __restrict__ idx0,
                              float* __restrict__ max0)
{
  __shared__ float sm[128];
  __shared__ int   si[128];
  const int b = blockIdx.x >> 9, n = blockIdx.x & 511, t = threadIdx.x;
  const float* row = C + ((size_t)b * NP1 + n) * NP1;
  const float un = u[b * NP1 + n];
  float best = -3.402823e38f; int bi = 0;
  for (int m = t; m < 512; m += 128) {
    const float val = row[m] + un + v[b * NP1 + m] + LOG1024f;
    if (val > best) { best = val; bi = m; }
  }
  sm[t] = best; si[t] = bi; __syncthreads();
  for (int o = 64; o > 0; o >>= 1) {
    if (t < o) {
      if (sm[t + o] > sm[t] || (sm[t + o] == sm[t] && si[t + o] < si[t])) {
        sm[t] = sm[t + o]; si[t] = si[t + o];
      }
    }
    __syncthreads();
  }
  if (t == 0) { idx0[b * Nc + n] = si[0]; max0[b * Nc + n] = sm[0]; }
}

// column argmax over n (u[n] term matters; v[m] is constant per column)
__global__ void colmax_kernel(const float* __restrict__ C, const float* __restrict__ u,
                              int* __restrict__ idx1)
{
  __shared__ float sm[128];
  __shared__ int   si[128];
  const int b = blockIdx.x >> 9, m = blockIdx.x & 511, t = threadIdx.x;
  const float* Cb = C + (size_t)b * NP1 * NP1;
  float best = -3.402823e38f; int bi = 0;
  for (int n = t; n < 512; n += 128) {
    const float val = Cb[(size_t)n * NP1 + m] + u[b * NP1 + n];
    if (val > best) { best = val; bi = n; }
  }
  sm[t] = best; si[t] = bi; __syncthreads();
  for (int o = 64; o > 0; o >>= 1) {
    if (t < o) {
      if (sm[t + o] > sm[t] || (sm[t + o] == sm[t] && si[t + o] < si[t])) {
        sm[t] = sm[t + o]; si[t] = si[t + o];
      }
    }
    __syncthreads();
  }
  if (t == 0) idx1[b * Nc + m] = si[0];
}

// out layout: [matches0 | matches1 | ms0 | ms1], each B*N floats
__global__ void match_a(const int* __restrict__ idx0, const int* __restrict__ idx1,
                        const float* __restrict__ max0, float* __restrict__ ms0buf,
                        float* __restrict__ val0buf, float* __restrict__ out)
{
  const int t = blockIdx.x * blockDim.x + threadIdx.x;
  if (t >= Bc * Nc) return;
  const int b = t / Nc, n = t % Nc;
  const int i0 = idx0[t];
  const bool mutual = (idx1[b * Nc + i0] == n);
  const float ms = mutual ? __expf(max0[t]) : 0.f;
  const bool valid = mutual && (ms > 0.2f);
  ms0buf[t] = ms;
  val0buf[t] = valid ? 1.f : 0.f;
  out[t] = valid ? (float)i0 : -1.f;                 // matches0
  out[(size_t)2 * Bc * Nc + t] = ms;                 // ms0
}

__global__ void match_b(const int* __restrict__ idx0, const int* __restrict__ idx1,
                        const float* __restrict__ ms0buf, const float* __restrict__ val0buf,
                        float* __restrict__ out)
{
  const int t = blockIdx.x * blockDim.x + threadIdx.x;
  if (t >= Bc * Nc) return;
  const int b = t / Nc, m = t % Nc;
  const int i1 = idx1[t];
  const bool mutual = (idx0[b * Nc + i1] == m);
  const float ms = mutual ? ms0buf[b * Nc + i1] : 0.f;
  const bool valid = mutual && (val0buf[b * Nc + i1] > 0.5f);
  out[(size_t)Bc * Nc + t] = valid ? (float)i1 : -1.f;   // matches1
  out[(size_t)3 * Bc * Nc + t] = ms;                     // ms1
}

// =====================================================================
// host side
// =====================================================================
static void launch_gemm(hipStream_t st, const float* W, const float* X, const float* bias,
                        const float* g, const float* be, const float* mu, const float* var,
                        const float* res, float* Y, int M, int K, int yld,
                        long xbs, long ybs, long rbs, int mode)
{
  dim3 grid(Nc / 64, M / 16, Bc);
  gemm_T<<<grid, dim3(32), 0, st>>>(W, X, bias, g, be, mu, var, res, Y,
                                    M, K, yld, xbs, ybs, rbs, mode);
}

static void propagate(hipStream_t st, const float* x, const float* src, float* out,
                      const float* Wq, const float* bq, const float* Wk, const float* bk,
                      const float* Wv, const float* bv, const float* Wm, const float* bm,
                      const float* W1, const float* b1, const float* g1, const float* be1,
                      const float* m1, const float* v1, const float* W2, const float* b2,
                      float* Qb, float* Kb, float* Vb, float* Tm, float* Cat, float* H1b,
                      float* Attn)
{
  const long DN = (long)Dc * Nc;
  // q,k: head-split layout; v: (c,n) layout
  launch_gemm(st, Wq, x,   bq, 0, 0, 0, 0, 0, Qb, Dc, Dc, 0,  DN, 0,  0, 3);
  launch_gemm(st, Wk, src, bk, 0, 0, 0, 0, 0, Kb, Dc, Dc, 0,  DN, 0,  0, 3);
  launch_gemm(st, Wv, src, bv, 0, 0, 0, 0, 0, Vb, Dc, Dc, 0,  DN, DN, 0, 4);
  attn_scores_wmma<<<dim3(Nc / 64, Nc / 16, Bc * Hc), dim3(32), 0, st>>>(Qb, Kb, Attn);
  softmax_rows<<<Bc * Hc * Nc, 128, 0, st>>>(Attn);
  attn_out_wmma<<<dim3(Nc / 16, 1, Bc * Hc), dim3(32), 0, st>>>(Attn, Vb, Tm);
  // message = Wm*attn_out + bm, into columns [256,512) of Cat_t
  launch_gemm(st, Wm, Tm, bm, 0, 0, 0, 0, 0, Cat + Dc, Dc, Dc, 2 * Dc, DN, 2 * DN, 0, 0);
  copy_to_cat<<<(Bc * Nc * Dc + 255) / 256, 256, 0, st>>>(x, Cat);
  launch_gemm(st, W1, Cat, b1, g1, be1, m1, v1, 0, H1b,
              2 * Dc, 2 * Dc, 2 * Dc, 2 * DN, 2 * DN, 0, 1);
  launch_gemm(st, W2, H1b, b2, 0, 0, 0, 0, x, out,
              Dc, 2 * Dc, Dc, 2 * DN, DN, DN, 2);
}

extern "C" void kernel_launch(void* const* d_in, const int* in_sizes, int n_in,
                              void* d_out, int out_size, void* d_ws, size_t ws_size,
                              hipStream_t stream)
{
  // ---- input mapping (pytree leaf order; None in enc_params drops out;
  //      gnn_params dict flattened with sorted keys) ----
  const float* kp0 = (const float*)d_in[0];
  const float* sc0 = (const float*)d_in[1];
  const float* de0 = (const float*)d_in[2];
  const float* kp1 = (const float*)d_in[3];
  const float* sc1 = (const float*)d_in[4];
  const float* de1 = (const float*)d_in[5];
  const float* E[26];
  for (int i = 0; i < 26; ++i) E[i] = (const float*)d_in[6 + i];
  // sorted keys: W1 W2 Wk Wm Wq Wv b1 b2 be1 bk bm bq bv g1 m1 v1
  const float* gW1 = (const float*)d_in[32];
  const float* gW2 = (const float*)d_in[33];
  const float* gWk = (const float*)d_in[34];
  const float* gWm = (const float*)d_in[35];
  const float* gWq = (const float*)d_in[36];
  const float* gWv = (const float*)d_in[37];
  const float* gb1 = (const float*)d_in[38];
  const float* gb2 = (const float*)d_in[39];
  const float* gbe1 = (const float*)d_in[40];
  const float* gbk = (const float*)d_in[41];
  const float* gbm = (const float*)d_in[42];
  const float* gbq = (const float*)d_in[43];
  const float* gbv = (const float*)d_in[44];
  const float* gg1 = (const float*)d_in[45];
  const float* gm1 = (const float*)d_in[46];
  const float* gv1 = (const float*)d_in[47];
  const float* finW = (const float*)d_in[48];
  const float* finB = (const float*)d_in[49];
  const float* alpha = (const float*)d_in[50];

  // ---- workspace layout (~51 MB of floats) ----
  float* ws = (float*)d_ws;
  const size_t S = (size_t)Bc * Dc * Nc;            // 524288
  float* X0 = ws;          float* X1 = ws + S;
  float* NX0 = ws + 2 * S; float* NX1 = ws + 3 * S;
  float* Qb = ws + 4 * S;  float* Kb = ws + 5 * S;  float* Vb = ws + 6 * S;
  float* Tm = ws + 7 * S;
  float* Cat = ws + 8 * S;                           // 2S
  float* H1b = ws + 10 * S;                          // 2S
  float* P0 = ws + 12 * S; float* P1 = ws + 13 * S;
  float* Attn = ws + 14 * S;                         // 8S  (B*H*N*N)
  float* Cpl = ws + 22 * S;                          // B*513*513
  float* Uv = Cpl + (size_t)Bc * NP1 * NP1;
  float* Vv = Uv + Bc * NP1;
  float* Max0 = Vv + Bc * NP1;
  float* Ms0 = Max0 + Bc * Nc;
  float* Val0 = Ms0 + Bc * Nc;
  int* I0 = (int*)(Val0 + Bc * Nc);
  int* I1 = I0 + Bc * Nc;

  // ---- keypoint encoder + residual add into transposed GNN state ----
  encoder_kernel<<<Bc * Nc, 256, 0, stream>>>(kp0, sc0, de0, X0,
      E[0], E[1], E[2], E[3], E[4], E[5],
      E[6], E[7], E[8], E[9], E[10], E[11],
      E[12], E[13], E[14], E[15], E[16], E[17],
      E[18], E[19], E[20], E[21], E[22], E[23],
      E[24], E[25]);
  encoder_kernel<<<Bc * Nc, 256, 0, stream>>>(kp1, sc1, de1, X1,
      E[0], E[1], E[2], E[3], E[4], E[5],
      E[6], E[7], E[8], E[9], E[10], E[11],
      E[12], E[13], E[14], E[15], E[16], E[17],
      E[18], E[19], E[20], E[21], E[22], E[23],
      E[24], E[25]);

  // ---- 18 GNN layers (self/cross alternating) ----
  float* cur0 = X0; float* cur1 = X1; float* alt0 = NX0; float* alt1 = NX1;
  const size_t DD = (size_t)Dc * Dc;
  for (int l = 0; l < Lc; ++l) {
    const float* Wq = gWq + (size_t)l * DD; const float* bq = gbq + (size_t)l * Dc;
    const float* Wk = gWk + (size_t)l * DD; const float* bk = gbk + (size_t)l * Dc;
    const float* Wv = gWv + (size_t)l * DD; const float* bv = gbv + (size_t)l * Dc;
    const float* Wm = gWm + (size_t)l * DD; const float* bm = gbm + (size_t)l * Dc;
    const float* W1 = gW1 + (size_t)l * (4 * DD); const float* b1 = gb1 + (size_t)l * 2 * Dc;
    const float* W2 = gW2 + (size_t)l * (2 * DD); const float* b2 = gb2 + (size_t)l * Dc;
    const float* g1 = gg1 + (size_t)l * 2 * Dc;  const float* be1 = gbe1 + (size_t)l * 2 * Dc;
    const float* m1 = gm1 + (size_t)l * 2 * Dc;  const float* v1 = gv1 + (size_t)l * 2 * Dc;
    const bool cross = (l & 1);
    const float* s0 = cross ? cur1 : cur0;
    const float* s1 = cross ? cur0 : cur1;
    propagate(stream, cur0, s0, alt0, Wq, bq, Wk, bk, Wv, bv, Wm, bm,
              W1, b1, g1, be1, m1, v1, W2, b2, Qb, Kb, Vb, Tm, Cat, H1b, Attn);
    propagate(stream, cur1, s1, alt1, Wq, bq, Wk, bk, Wv, bv, Wm, bm,
              W1, b1, g1, be1, m1, v1, W2, b2, Qb, Kb, Vb, Tm, Cat, H1b, Attn);
    float* t0 = cur0; cur0 = alt0; alt0 = t0;
    float* t1 = cur1; cur1 = alt1; alt1 = t1;
  }

  // ---- final projection + match scores + Sinkhorn + matching ----
  const long DN = (long)Dc * Nc;
  launch_gemm(stream, finW, cur0, finB, 0, 0, 0, 0, 0, P0, Dc, Dc, Dc, DN, DN, 0, 0);
  launch_gemm(stream, finW, cur1, finB, 0, 0, 0, 0, 0, P1, Dc, Dc, Dc, DN, DN, 0, 0);
  final_scores_wmma<<<dim3(Nc / 64, Nc / 16, Bc), dim3(32), 0, stream>>>(P0, P1, Cpl);
  bins_kernel<<<(Bc * NP1 + 255) / 256, 256, 0, stream>>>(Cpl, alpha);
  sinkhorn_kernel<<<Bc, 512, 0, stream>>>(Cpl, Uv, Vv);
  rowmax_kernel<<<Bc * Nc, 128, 0, stream>>>(Cpl, Uv, Vv, I0, Max0);
  colmax_kernel<<<Bc * Nc, 128, 0, stream>>>(Cpl, Uv, I1);
  match_a<<<(Bc * Nc + 255) / 256, 256, 0, stream>>>(I0, I1, Max0, Ms0, Val0, (float*)d_out);
  match_b<<<(Bc * Nc + 255) / 256, 256, 0, stream>>>(I0, I1, Ms0, Val0, (float*)d_out);
}
